// TwoLayerNet_45518063403579
// MI455X (gfx1250) — matrix operand order repaired and checked by reading, not compiled, not run on hardware
//
#include <hip/hip_runtime.h>

#define B_ROWS 4096
#define D_IN   64
#define H_DIM  128
#define D_OUT  512
#define K_TOT  (D_IN * H_DIM + H_DIM)   // 8320 = Km*u part + e part
#define NCHUNK (K_TOT / 32)             // 260 K-chunks of 32
#define NT_TOT (D_OUT / 16)             // 32 N-tiles of 16
#define LFS    72                        // padded LF row stride (halfs), [64] == 1.0

typedef __attribute__((ext_vector_type(16))) _Float16 v16h;
typedef __attribute__((ext_vector_type(8)))  _Float16 v8h;
typedef __attribute__((ext_vector_type(8)))  float    v8f;
typedef __attribute__((ext_vector_type(4)))  int      v4i;

#if defined(__has_builtin)
#if __has_builtin(__builtin_amdgcn_global_load_async_to_lds_b128)
#define USE_ASYNC_B 1
#endif
#endif
#ifndef USE_ASYNC_B
#define USE_ASYNC_B 0
#endif

#define GLB1 __attribute__((address_space(1)))
#define LDS3 __attribute__((address_space(3)))

// CDNA5 split-counter wave-local LDS fence: DS ops are in-order per wave,
// s_wait_dscnt 0 + memory clobber makes cross-lane LDS RAW safe without a
// block barrier (waves run divergent fixed-point iteration counts).
__device__ __forceinline__ void wave_lds_fence() {
  asm volatile("s_wait_dscnt 0" ::: "memory");
}

// ---------------------------------------------------------------------------
// Prep: build the GEMM B operand pre-swizzled into WMMA 16-bit B layout.
// B[k, n] for k < 8192: relu(K)[i,h] * u[k, n]  (k = i*128 + h)
//          k >= 8192 : e[k-8192, n]             (folds the AF@e term)
// Layout: flat[((kc*NT_TOT + nt)*32 + lane)*16 + t]
//   n    = nt*16 + (lane&15)
//   K    = kc*32 + ((lane&16) ? 8 : 0) + (t < 8 ? t : t + 8)  (t-th half of the v16h operand)
// => each GEMM B fragment is 32 contiguous bytes per lane.
// ---------------------------------------------------------------------------
__global__ void __launch_bounds__(256) prep_kernel(
    const float* __restrict__ Kg, const float* __restrict__ u,
    const float* __restrict__ e, _Float16* __restrict__ Bsw)
{
  const int idx  = blockIdx.x * 256 + threadIdx.x;
  const int t    = idx & 15;
  const int lane = (idx >> 4) & 31;
  const int nt   = (idx >> 9) & 31;
  const int kc   = idx >> 14;
  if (kc >= NCHUNK) return;
  const int n = nt * 16 + (lane & 15);
  const int k = kc * 32 + ((lane & 16) ? 8 : 0) + (t < 8 ? t : t + 8);
  float val;
  if (k < D_IN * H_DIM) {
    val = fmaxf(Kg[k], 0.0f) * u[(size_t)k * D_OUT + n];
  } else {
    val = e[(size_t)(k - D_IN * H_DIM) * D_OUT + n];
  }
  Bsw[idx] = (_Float16)val;
}

// ---------------------------------------------------------------------------
// Fixed-point solver. 128 threads = 4 waves; each wave owns 4 batch rows.
// K (relu'd) and its transpose live in LDS (f32); AF/LF/TMP vectors live in
// per-wave LDS scratch so every lane can read the full vectors with b128 loads.
// Per iteration per row:  Ldi = LT/(K@AF+1); AF' = AT/(KT@Ldi+1);
//                         Adi = AT/(KT@LF+1); LF' = LT/(K@Adi+1)
// Convergence: max rel-diff over the wave's rows (shfl_xor reduction), then
// one extra update (the reference's post-loop Km pass) and f16 store of AF/LF.
// ---------------------------------------------------------------------------
__global__ void __launch_bounds__(128) solver_kernel(
    const float* __restrict__ LT, const float* __restrict__ Kg,
    const float* __restrict__ ATg, const int* __restrict__ max_step_p,
    _Float16* __restrict__ AFws, _Float16* __restrict__ LFws)
{
  constexpr int SW  = 4;  // waves per block
  constexpr int RPW = 4;  // rows per wave
  __shared__ float Ks [D_IN * H_DIM];
  __shared__ float KTs[H_DIM * D_IN];
  __shared__ float ATs[H_DIM];
  __shared__ float AFl [SW][RPW][H_DIM];
  __shared__ float LFl [SW][RPW][D_IN];
  __shared__ float TMPl[SW][RPW][H_DIM];

  const int tid = threadIdx.x;
  for (int x = tid; x < D_IN * H_DIM; x += 128) {
    const float km = fmaxf(Kg[x], 0.0f);
    Ks[x] = km;
    KTs[(x & (H_DIM - 1)) * D_IN + (x >> 7)] = km;
  }
  if (tid < H_DIM) ATs[tid] = fmaxf(ATg[tid], 0.0f);
  __syncthreads();

  const int w    = tid >> 5;
  const int lane = tid & 31;
  const int max_step = max_step_p[0];
  const int row0 = (blockIdx.x * SW + w) * RPW;

  float lt0[RPW], lt1[RPW];
  #pragma unroll
  for (int r = 0; r < RPW; ++r) {
    lt0[r] = LT[(size_t)(row0 + r) * D_IN + lane];
    lt1[r] = LT[(size_t)(row0 + r) * D_IN + lane + 32];
  }
  const float at0 = ATs[lane], at1 = ATs[lane + 32];
  const float at2 = ATs[lane + 64], at3 = ATs[lane + 96];

  float af[RPW][4], lf[RPW][2];
  #pragma unroll
  for (int r = 0; r < RPW; ++r) {
    af[r][0] = af[r][1] = af[r][2] = af[r][3] = 0.0f;
    lf[r][0] = lf[r][1] = 0.0f;
    AFl[w][r][lane] = 0.0f; AFl[w][r][lane + 32] = 0.0f;
    AFl[w][r][lane + 64] = 0.0f; AFl[w][r][lane + 96] = 0.0f;
    LFl[w][r][lane] = 0.0f; LFl[w][r][lane + 32] = 0.0f;
  }
  wave_lds_fence();

  auto body = [&](bool want_diff, float& dA, float& dL) {
    // ---- Phase A: Ldi = LT/(K@AF+1)  ->  AF' = AT/(KT@Ldi+1) ----
    {
      float s0[RPW] = {}, s1[RPW] = {};
      for (int h = 0; h < H_DIM; h += 4) {
        const float4 k0 = *(const float4*)&Ks[lane * H_DIM + h];
        const float4 k1 = *(const float4*)&Ks[(lane + 32) * H_DIM + h];
        #pragma unroll
        for (int r = 0; r < RPW; ++r) {
          const float4 av = *(const float4*)&AFl[w][r][h];
          s0[r] += k0.x*av.x + k0.y*av.y + k0.z*av.z + k0.w*av.w;
          s1[r] += k1.x*av.x + k1.y*av.y + k1.z*av.z + k1.w*av.w;
        }
      }
      #pragma unroll
      for (int r = 0; r < RPW; ++r) {
        TMPl[w][r][lane]      = lt0[r] / (s0[r] + 1.0f);
        TMPl[w][r][lane + 32] = lt1[r] / (s1[r] + 1.0f);
      }
      wave_lds_fence();
      float t0[RPW] = {}, t1[RPW] = {}, t2[RPW] = {}, t3[RPW] = {};
      for (int i = 0; i < D_IN; i += 4) {
        const float4 c0 = *(const float4*)&KTs[lane * D_IN + i];
        const float4 c1 = *(const float4*)&KTs[(lane + 32) * D_IN + i];
        const float4 c2 = *(const float4*)&KTs[(lane + 64) * D_IN + i];
        const float4 c3 = *(const float4*)&KTs[(lane + 96) * D_IN + i];
        #pragma unroll
        for (int r = 0; r < RPW; ++r) {
          const float4 dv = *(const float4*)&TMPl[w][r][i];
          t0[r] += c0.x*dv.x + c0.y*dv.y + c0.z*dv.z + c0.w*dv.w;
          t1[r] += c1.x*dv.x + c1.y*dv.y + c1.z*dv.z + c1.w*dv.w;
          t2[r] += c2.x*dv.x + c2.y*dv.y + c2.z*dv.z + c2.w*dv.w;
          t3[r] += c3.x*dv.x + c3.y*dv.y + c3.z*dv.z + c3.w*dv.w;
        }
      }
      #pragma unroll
      for (int r = 0; r < RPW; ++r) {
        const float n0 = at0 / (t0[r] + 1.0f), n1 = at1 / (t1[r] + 1.0f);
        const float n2 = at2 / (t2[r] + 1.0f), n3 = at3 / (t3[r] + 1.0f);
        if (want_diff) {
          dA = fmaxf(dA, fabsf(n0 - af[r][0]) / (af[r][0] + 1e-5f));
          dA = fmaxf(dA, fabsf(n1 - af[r][1]) / (af[r][1] + 1e-5f));
          dA = fmaxf(dA, fabsf(n2 - af[r][2]) / (af[r][2] + 1e-5f));
          dA = fmaxf(dA, fabsf(n3 - af[r][3]) / (af[r][3] + 1e-5f));
        }
        af[r][0] = n0; af[r][1] = n1; af[r][2] = n2; af[r][3] = n3;
        AFl[w][r][lane] = n0; AFl[w][r][lane + 32] = n1;
        AFl[w][r][lane + 64] = n2; AFl[w][r][lane + 96] = n3;
      }
      wave_lds_fence();
    }
    // ---- Phase L: Adi = AT/(KT@LF+1)  ->  LF' = LT/(K@Adi+1) ----
    {
      float t0[RPW] = {}, t1[RPW] = {}, t2[RPW] = {}, t3[RPW] = {};
      for (int i = 0; i < D_IN; i += 4) {
        const float4 c0 = *(const float4*)&KTs[lane * D_IN + i];
        const float4 c1 = *(const float4*)&KTs[(lane + 32) * D_IN + i];
        const float4 c2 = *(const float4*)&KTs[(lane + 64) * D_IN + i];
        const float4 c3 = *(const float4*)&KTs[(lane + 96) * D_IN + i];
        #pragma unroll
        for (int r = 0; r < RPW; ++r) {
          const float4 lv = *(const float4*)&LFl[w][r][i];
          t0[r] += c0.x*lv.x + c0.y*lv.y + c0.z*lv.z + c0.w*lv.w;
          t1[r] += c1.x*lv.x + c1.y*lv.y + c1.z*lv.z + c1.w*lv.w;
          t2[r] += c2.x*lv.x + c2.y*lv.y + c2.z*lv.z + c2.w*lv.w;
          t3[r] += c3.x*lv.x + c3.y*lv.y + c3.z*lv.z + c3.w*lv.w;
        }
      }
      #pragma unroll
      for (int r = 0; r < RPW; ++r) {
        TMPl[w][r][lane]      = at0 / (t0[r] + 1.0f);
        TMPl[w][r][lane + 32] = at1 / (t1[r] + 1.0f);
        TMPl[w][r][lane + 64] = at2 / (t2[r] + 1.0f);
        TMPl[w][r][lane + 96] = at3 / (t3[r] + 1.0f);
      }
      wave_lds_fence();
      float s0[RPW] = {}, s1[RPW] = {};
      for (int h = 0; h < H_DIM; h += 4) {
        const float4 k0 = *(const float4*)&Ks[lane * H_DIM + h];
        const float4 k1 = *(const float4*)&Ks[(lane + 32) * H_DIM + h];
        #pragma unroll
        for (int r = 0; r < RPW; ++r) {
          const float4 av = *(const float4*)&TMPl[w][r][h];
          s0[r] += k0.x*av.x + k0.y*av.y + k0.z*av.z + k0.w*av.w;
          s1[r] += k1.x*av.x + k1.y*av.y + k1.z*av.z + k1.w*av.w;
        }
      }
      #pragma unroll
      for (int r = 0; r < RPW; ++r) {
        const float n0 = lt0[r] / (s0[r] + 1.0f);
        const float n1 = lt1[r] / (s1[r] + 1.0f);
        if (want_diff) {
          dL = fmaxf(dL, fabsf(n0 - lf[r][0]) / (lf[r][0] + 1e-5f));
          dL = fmaxf(dL, fabsf(n1 - lf[r][1]) / (lf[r][1] + 1e-5f));
        }
        lf[r][0] = n0; lf[r][1] = n1;
        LFl[w][r][lane] = n0; LFl[w][r][lane + 32] = n1;
      }
      wave_lds_fence();
    }
  };

  float diff = 1e9f;
  int step = 0;
  while (step < max_step && diff >= 1e-3f) {
    float dA = 0.0f, dL = 0.0f;
    body(true, dA, dL);
    #pragma unroll
    for (int off = 16; off > 0; off >>= 1) {
      dA = fmaxf(dA, __shfl_xor(dA, off));
      dL = fmaxf(dL, __shfl_xor(dL, off));
    }
    diff = dA + dL;
    ++step;
  }
  float du0 = 0.0f, du1 = 0.0f;
  body(false, du0, du1);  // reference's post-loop Km refresh (same update)

  #pragma unroll
  for (int r = 0; r < RPW; ++r) {
    const int row = row0 + r;
    _Float16* afp = AFws + (size_t)row * H_DIM;
    afp[lane]      = (_Float16)af[r][0];
    afp[lane + 32] = (_Float16)af[r][1];
    afp[lane + 64] = (_Float16)af[r][2];
    afp[lane + 96] = (_Float16)af[r][3];
    _Float16* lfp = LFws + (size_t)row * LFS;
    lfp[lane]      = (_Float16)lf[r][0];
    lfp[lane + 32] = (_Float16)lf[r][1];
    if (lane == 0) lfp[D_IN] = (_Float16)1.0f;  // identity row for the e-term
  }
}

// ---------------------------------------------------------------------------
// Fused GEMM: out[b,n] = sum_k G[b,k]*Bsw[k,n] + bias[n],  K = 8320
//   G[b,k] = LF[b, k>>7] * AF[b, k&127]   (LF[b,64]==1 covers the e-rows)
// Block: 128 threads (4 wave32), tile 64(M) x 128(N); per wave 16x128.
// B path: per chunk, the block stages one 8 KB slab of the pre-swizzled Bsw
// into LDS ONCE via GLOBAL_LOAD_ASYNC_TO_LDS_B128 (double-buffered; the copy
// of chunk kc+1 overlaps the 8 WMMAs of chunk kc). Sync: s_wait_asynccnt 0
// + barrier (staging done), s_wait_dscnt 0 + barrier (readers done).
// Fallback (builtin absent): direct global loads, hoisted ahead of the WMMAs.
// ---------------------------------------------------------------------------
__global__ void __launch_bounds__(128) gemm_kernel(
    const _Float16* __restrict__ Bsw, const _Float16* __restrict__ AFws,
    const _Float16* __restrict__ LFws, const float* __restrict__ bias,
    float* __restrict__ out)
{
  const int tid   = threadIdx.x;
  const int w     = tid >> 5;
  const int lane  = tid & 31;
  const int mbase = blockIdx.x * 64 + w * 16;
  const int nblk  = blockIdx.y;
  const int halfSel = (lane & 16) ? 8 : 0;            // A-layout K sub-run select
  const int mrow  = mbase + (lane & 15);
  const _Float16* __restrict__ afrow = AFws + (size_t)mrow * H_DIM;
  const _Float16* __restrict__ lfrow = LFws + (size_t)mrow * LFS;

  v8f acc[8] = {};

#if USE_ASYNC_B
  __shared__ _Float16 lbuf[2][8 * 32 * 16];  // 2 x 8 KB K-chunk slabs

  auto stage = [&](int kc) {
    const int buf = kc & 1;
    const _Float16* src = Bsw + ((size_t)kc * NT_TOT + nblk * 8) * (32 * 16);
    #pragma unroll
    for (int q = 0; q < 4; ++q) {
      const int unit = tid + 128 * q;  // 512 x 16-byte units = 8 KB
      __builtin_amdgcn_global_load_async_to_lds_b128(
          (GLB1 v4i*)(void*)(src + (size_t)unit * 8),
          (LDS3 v4i*)(void*)(&lbuf[buf][unit * 8]),
          0, 0);
    }
  };

  stage(0);
  for (int kc = 0; kc < NCHUNK; ++kc) {
    const int k0 = kc * 32;
    const int h0 = k0 & (H_DIM - 1);
    const _Float16 lfv = lfrow[k0 >> 7];
    const v8h a0 = *(const v8h*)(afrow + h0 + halfSel);
    const v8h a1 = *(const v8h*)(afrow + h0 + 16 + halfSel);
    v16h a;
    #pragma unroll
    for (int t = 0; t < 8; ++t) { a[t] = a0[t] * lfv; a[8 + t] = a1[t] * lfv; }

    asm volatile("s_wait_asynccnt 0" ::: "memory");   // our staging landed
    __syncthreads();                                   // whole block's staging landed
    const _Float16* lb = &lbuf[kc & 1][lane * 16];
    v16h bfrag[8];
    #pragma unroll
    for (int j = 0; j < 8; ++j) bfrag[j] = *(const v16h*)(lb + j * 512);
    asm volatile("s_wait_dscnt 0" ::: "memory");       // fragment reads complete
    __syncthreads();                                   // whole block done reading
    if (kc + 1 < NCHUNK) stage(kc + 1);                // DMA overlaps the WMMAs

    #pragma unroll
    for (int j = 0; j < 8; ++j) {
      acc[j] = __builtin_amdgcn_wmma_f32_16x16x32_f16(
          /*neg_a=*/false, a, /*neg_b=*/false, bfrag[j],
          /*c_mod=*/(short)0, acc[j], /*reuse_a=*/false, /*reuse_b=*/false);
    }
  }
#else
  for (int kc = 0; kc < NCHUNK; ++kc) {
    const int k0 = kc * 32;
    const int h0 = k0 & (H_DIM - 1);
    const _Float16 lfv = lfrow[k0 >> 7];
    const v8h a0 = *(const v8h*)(afrow + h0 + halfSel);
    const v8h a1 = *(const v8h*)(afrow + h0 + 16 + halfSel);
    v16h a;
    #pragma unroll
    for (int t = 0; t < 8; ++t) { a[t] = a0[t] * lfv; a[8 + t] = a1[t] * lfv; }

    const _Float16* bkc = Bsw + (((size_t)kc * NT_TOT + nblk * 8) * 32 + lane) * 16;
    __builtin_prefetch((const void*)(bkc + (size_t)NT_TOT * 32 * 16), 0, 3);
    v16h bfrag[8];
    #pragma unroll
    for (int j = 0; j < 8; ++j) bfrag[j] = *(const v16h*)(bkc + (size_t)j * 512);
    #pragma unroll
    for (int j = 0; j < 8; ++j) {
      acc[j] = __builtin_amdgcn_wmma_f32_16x16x32_f16(
          false, a, false, bfrag[j], (short)0, acc[j], false, false);
    }
  }
#endif

  // C/D layout: VGPR r: lanes 0-15 -> (M=mbase+r, N=lane), lanes 16-31 -> (M=mbase+8+r)
  const int ncb = nblk * 128 + (lane & 15);
  const int mw  = mbase + ((lane & 16) ? 8 : 0);
  #pragma unroll
  for (int j = 0; j < 8; ++j) {
    const int n = ncb + j * 16;
    const float bb = bias[n];
    #pragma unroll
    for (int r = 0; r < 8; ++r) {
      out[(size_t)(mw + r) * D_OUT + n] = acc[j][r] + bb;
    }
  }
}

extern "C" void kernel_launch(void* const* d_in, const int* in_sizes, int n_in,
                              void* d_out, int out_size, void* d_ws, size_t ws_size,
                              hipStream_t stream) {
  const float* LT   = (const float*)d_in[0];
  const float* Kg   = (const float*)d_in[1];
  const float* AT   = (const float*)d_in[2];
  const float* u    = (const float*)d_in[3];
  const float* e    = (const float*)d_in[4];
  const float* bias = (const float*)d_in[5];
  const int*   ms   = (const int*)d_in[6];
  float* out = (float*)d_out;

  char* ws = (char*)d_ws;
  _Float16* Bsw = (_Float16*)ws;                                   // 8,519,680 B
  const size_t bsw_bytes = (size_t)NCHUNK * NT_TOT * 32 * 16 * 2;
  _Float16* AFws = (_Float16*)(ws + bsw_bytes);                    // 1,048,576 B
  const size_t af_bytes = (size_t)B_ROWS * H_DIM * 2;
  _Float16* LFws = (_Float16*)(ws + bsw_bytes + af_bytes);         //   589,824 B

  prep_kernel<<<dim3((NCHUNK * NT_TOT * 32 * 16) / 256), 256, 0, stream>>>(Kg, u, e, Bsw);
  solver_kernel<<<dim3(B_ROWS / 16), 128, 0, stream>>>(LT, Kg, AT, ms, AFws, LFws);
  gemm_kernel<<<dim3(B_ROWS / 64, D_OUT / 128), 128, 0, stream>>>(Bsw, AFws, LFws, bias, out);
}
